// HierNetExp_81527069212969
// MI455X (gfx1250) — matrix operand order repaired and checked by reading, not compiled, not run on hardware
//
#include <hip/hip_runtime.h>
#include <hip/hip_bf16.h>

#define NN 16
#define BB 1024
#define FF 1024
#define TT 128
#define CC 2048
#define MASK_EPS 1e-17f

typedef __attribute__((ext_vector_type(16))) __bf16 v16bf;
typedef __attribute__((ext_vector_type(8)))  __bf16 v8bf;
typedef __attribute__((ext_vector_type(8)))  float  v8f;

union frag16 { v16bf v; v8bf h[2]; };

// ---------------- conversion kernels ----------------

// plain f32 -> bf16 convert (x)
__global__ void conv_bf16(const float* __restrict__ in, __bf16* __restrict__ out, size_t n) {
    size_t idx = (size_t)blockIdx.x * blockDim.x + threadIdx.x;
    size_t stride = (size_t)gridDim.x * blockDim.x;
    for (; idx < n; idx += stride) out[idx] = (__bf16)in[idx];
}

// transpose-convert: in [R, C] f32 (row-major) -> out [C, R] bf16, scaled.
// grid: x = C/128, y = R/32, z = batch.  tri==true: z encodes (i,j), skip j>i.
__global__ __launch_bounds__(256) void convT(const float* __restrict__ in,
                                             __bf16* __restrict__ out,
                                             int R, int C, float scale, int tri) {
    if (tri) {
        int i = blockIdx.z >> 4, j = blockIdx.z & 15;
        if (j > i) return;
    }
    in  += (size_t)blockIdx.z * R * C;
    out += (size_t)blockIdx.z * R * C;
    int r0 = blockIdx.y * 32, c0 = blockIdx.x * 128;
    __shared__ __bf16 tile[32][136];
    int tid = threadIdx.x;
#pragma unroll
    for (int it = 0; it < 16; ++it) {
        int idx = it * 256 + tid;
        int r = idx >> 7, c = idx & 127;
        tile[r][c] = (__bf16)(in[(size_t)(r0 + r) * C + c0 + c] * scale);
    }
    __syncthreads();
    int c = tid & 127, h = tid >> 7;   // h = 0..1, each writes 16 consecutive r
    v8bf v0, v1;
#pragma unroll
    for (int e = 0; e < 8; ++e) {
        v0[e] = tile[h * 16 + e][c];
        v1[e] = tile[h * 16 + 8 + e][c];
    }
    v8bf* dst = (v8bf*)(out + (size_t)(c0 + c) * R + r0 + h * 16);
    dst[0] = v0;
    dst[1] = v1;
}

// bm[i,t] = 0.2 * sum_{j<=i} b[i,j,t]
__global__ void bias_reduce(const float* __restrict__ b, float* __restrict__ bm) {
    int i = blockIdx.x, t = threadIdx.x;
    float s = 0.f;
    for (int j = 0; j <= i; ++j) s += b[((size_t)i * NN + j) * TT + t];
    bm[i * TT + t] = 0.2f * s;
}

// ---------------- GEMM 1: nout ----------------
// nout[i, m, t] = sum_{j<=i} xh[j, m, :] @ wt[i,j][t, :]  + bm[i,t]
// xh: [N, B, F] bf16 row-major ; wt: [N, N, T, F] bf16 (pre-scaled by 0.2)
// grid: x = B/128 m-tiles, y = node i ; block 256 (8 waves)
__global__ __launch_bounds__(256) void gemm1(const __bf16* __restrict__ xh,
                                             const __bf16* __restrict__ wt,
                                             const float*  __restrict__ bm,
                                             float* __restrict__ nout) {
    const int i    = blockIdx.y;
    const int m0   = blockIdx.x * 128;
    const int wave = threadIdx.x >> 5;
    const int lane = threadIdx.x & 31;
    const int wm   = wave >> 1;      // 0..3 : 32-row strip
    const int wn   = wave & 1;       // 0..1 : 64-col strip
    const int laneHi = lane >> 4;
    const int lane15 = lane & 15;

    v8f acc[2][4];
#pragma unroll
    for (int s = 0; s < 2; ++s)
#pragma unroll
        for (int q = 0; q < 4; ++q) acc[s][q] = (v8f){0.f,0.f,0.f,0.f,0.f,0.f,0.f,0.f};

    for (int j = 0; j <= i; ++j) {
        const __bf16* A  = xh + (size_t)j * BB * FF;
        const __bf16* Bw = wt + (((size_t)i * NN + j) * TT) * FF;
        for (int kk = 0; kk < FF; kk += 32) {
            // prefetch next K block (global_prefetch_b8 path)
            if (kk + 64 < FF) {
                __builtin_prefetch(A + (size_t)(m0 + wm * 32 + lane15) * FF + kk + 64, 0, 0);
                __builtin_prefetch(Bw + (size_t)(wn * 64 + lane15) * FF + kk + 64, 0, 0);
            }
            frag16 af[2];
#pragma unroll
            for (int s = 0; s < 2; ++s) {
                int row = m0 + wm * 32 + s * 16 + lane15;
                const v8bf* p = (const v8bf*)(A + (size_t)row * FF + kk + laneHi * 8);
                af[s].h[0] = p[0];      // K 0..7   (lanes>=16: 8..15)
                af[s].h[1] = p[2];      // K 16..23 (lanes>=16: 24..31)
            }
            frag16 bfr[4];
#pragma unroll
            for (int q = 0; q < 4; ++q) {
                int t = wn * 64 + q * 16 + lane15;
                const v8bf* p = (const v8bf*)(Bw + (size_t)t * FF + kk + laneHi * 8);
                bfr[q].h[0] = p[0];     // K 0..7   (lanes>=16: 8..15)
                bfr[q].h[1] = p[2];     // K 16..23 (lanes>=16: 24..31)
            }
#pragma unroll
            for (int s = 0; s < 2; ++s)
#pragma unroll
                for (int q = 0; q < 4; ++q)
                    acc[s][q] = __builtin_amdgcn_wmma_f32_16x16x32_bf16(
                        false, af[s].v, false, bfr[q].v, (short)0, acc[s][q], false, false);
        }
    }
#pragma unroll
    for (int s = 0; s < 2; ++s)
#pragma unroll
        for (int q = 0; q < 4; ++q) {
            int t = wn * 64 + q * 16 + lane15;
            float bb = bm[i * TT + t];
#pragma unroll
            for (int r = 0; r < 8; ++r) {
                int row = m0 + wm * 32 + s * 16 + laneHi * 8 + r;
                nout[((size_t)i * BB + row) * TT + t] = acc[s][q][r] + bb;
            }
        }
}

// ---------------- weight nout by inclusive gate cumprod ----------------
// wnout[b, i*T+t] = (prod_{p<=i} gate[b,p]) * nout[i,b,t]   (bf16)
__global__ void build_wnout(const float* __restrict__ nout,
                            const float* __restrict__ gate,
                            __bf16* __restrict__ wnout) {
    int b = blockIdx.x;
    __shared__ float w[NN];
    if (threadIdx.x < NN) {
        float p = 1.f;
        for (int q = 0; q <= (int)threadIdx.x; ++q) p *= gate[(size_t)b * NN + q];
        w[threadIdx.x] = p;
    }
    __syncthreads();
    for (int idx = threadIdx.x; idx < NN * TT; idx += 256) {
        int i = idx >> 7, t = idx & 127;
        wnout[(size_t)b * (NN * TT) + idx] =
            (__bf16)(w[i] * nout[((size_t)i * BB + b) * TT + t]);
    }
}

// ---------------- GEMM 2: output ----------------
// output[m, c] = wnout[m, :] @ cwt[c, :]   (K = N*T = 2048)
// grid: x = B/128, y = C/128 ; block 256 (8 waves)
__global__ __launch_bounds__(256) void gemm2(const __bf16* __restrict__ A,
                                             const __bf16* __restrict__ Bt,
                                             float* __restrict__ out) {
    const int K = NN * TT;           // 2048
    const int m0 = blockIdx.x * 128;
    const int n0 = blockIdx.y * 128;
    const int wave = threadIdx.x >> 5;
    const int lane = threadIdx.x & 31;
    const int wm = wave >> 1, wn = wave & 1;
    const int laneHi = lane >> 4, lane15 = lane & 15;

    v8f acc[2][4];
#pragma unroll
    for (int s = 0; s < 2; ++s)
#pragma unroll
        for (int q = 0; q < 4; ++q) acc[s][q] = (v8f){0.f,0.f,0.f,0.f,0.f,0.f,0.f,0.f};

    for (int kk = 0; kk < K; kk += 32) {
        if (kk + 64 < K) {
            __builtin_prefetch(A + (size_t)(m0 + wm * 32 + lane15) * K + kk + 64, 0, 0);
            __builtin_prefetch(Bt + (size_t)(n0 + wn * 64 + lane15) * K + kk + 64, 0, 0);
        }
        frag16 af[2];
#pragma unroll
        for (int s = 0; s < 2; ++s) {
            int row = m0 + wm * 32 + s * 16 + lane15;
            const v8bf* p = (const v8bf*)(A + (size_t)row * K + kk + laneHi * 8);
            af[s].h[0] = p[0];
            af[s].h[1] = p[2];
        }
        frag16 bfr[4];
#pragma unroll
        for (int q = 0; q < 4; ++q) {
            int c = n0 + wn * 64 + q * 16 + lane15;
            const v8bf* p = (const v8bf*)(Bt + (size_t)c * K + kk + laneHi * 8);
            bfr[q].h[0] = p[0];
            bfr[q].h[1] = p[2];
        }
#pragma unroll
        for (int s = 0; s < 2; ++s)
#pragma unroll
            for (int q = 0; q < 4; ++q)
                acc[s][q] = __builtin_amdgcn_wmma_f32_16x16x32_bf16(
                    false, af[s].v, false, bfr[q].v, (short)0, acc[s][q], false, false);
    }
#pragma unroll
    for (int s = 0; s < 2; ++s)
#pragma unroll
        for (int q = 0; q < 4; ++q) {
            int c = n0 + wn * 64 + q * 16 + lane15;
#pragma unroll
            for (int r = 0; r < 8; ++r) {
                int row = m0 + wm * 32 + s * 16 + laneHi * 8 + r;
                out[(size_t)row * CC + c] = acc[s][q][r];
            }
        }
}

// ---------------- epilogue: mask + exp reduction ----------------
__global__ void epilogue(const float* __restrict__ output,
                         const float* __restrict__ gate,
                         const float* __restrict__ node_mask,
                         float* __restrict__ sfmx) {
    int b = blockIdx.x;
    __shared__ float og[NN];
    __shared__ float red[256];
    if (threadIdx.x < NN) og[threadIdx.x] = 1.f - gate[(size_t)b * NN + threadIdx.x];
    __syncthreads();
    float acc = 0.f;
    for (int c = threadIdx.x; c < CC; c += 256) {
        float ms = 0.f;
#pragma unroll
        for (int i = 0; i < NN; ++i) {
            float m = node_mask[(size_t)i * CC + c];
            m = m < MASK_EPS ? MASK_EPS : (m > 1.f ? 1.f : m);
            ms += og[i] * __logf(m);
        }
        if (ms == 0.f) acc += __expf(output[(size_t)b * CC + c]);
    }
    red[threadIdx.x] = acc;
    __syncthreads();
    for (int s = 128; s > 0; s >>= 1) {
        if ((int)threadIdx.x < s) red[threadIdx.x] += red[threadIdx.x + s];
        __syncthreads();
    }
    if (threadIdx.x == 0) sfmx[b] = red[0];
}

// ---------------- launch ----------------
extern "C" void kernel_launch(void* const* d_in, const int* in_sizes, int n_in,
                              void* d_out, int out_size, void* d_ws, size_t ws_size,
                              hipStream_t stream) {
    const float* x    = (const float*)d_in[0];   // [N,B,F]
    const float* W    = (const float*)d_in[1];   // [N,N,F,T]
    const float* b    = (const float*)d_in[2];   // [N,N,T]
    const float* cw   = (const float*)d_in[3];   // [N,T,C]
    const float* nm   = (const float*)d_in[4];   // [N,C]
    const float* gate = (const float*)d_in[5];   // [B,N]

    float* out_output = (float*)d_out;                              // [B,C]
    float* out_nout   = out_output + (size_t)BB * CC;               // [N,B,T]
    float* out_sfmx   = out_nout + (size_t)NN * BB * TT;            // [B]

    char* ws = (char*)d_ws;
    __bf16* xh  = (__bf16*)ws;  ws += (size_t)NN * BB * FF * 2;        // 32 MB
    __bf16* wt  = (__bf16*)ws;  ws += (size_t)NN * NN * TT * FF * 2;   // 67 MB
    __bf16* cwt = (__bf16*)ws;  ws += (size_t)CC * (NN * TT) * 2;      // 8 MB
    __bf16* wn  = (__bf16*)ws;  ws += (size_t)BB * (NN * TT) * 2;      // 4 MB
    float*  bm  = (float*)ws;                                          // 8 KB

    // 1) x -> bf16
    conv_bf16<<<2048, 256, 0, stream>>>(x, xh, (size_t)NN * BB * FF);
    // 2) W[i,j] (FxT, scaled 1/5) -> wt[i,j] (TxF) bf16, only j<=i
    convT<<<dim3(TT / 128, FF / 32, NN * NN), 256, 0, stream>>>(W, wt, FF, TT, 0.2f, 1);
    // 3) cw [(i,t) x c] -> cwt [c x (i,t)] bf16
    convT<<<dim3(CC / 128, (NN * TT) / 32, 1), 256, 0, stream>>>(cw, cwt, NN * TT, CC, 1.0f, 0);
    // 4) bias reduction with 1/5 fold
    bias_reduce<<<NN, TT, 0, stream>>>(b, bm);
    // 5) big triangular GEMM -> nout (f32 output)
    gemm1<<<dim3(BB / 128, NN), 256, 0, stream>>>(xh, wt, bm, out_nout);
    // 6) gate cumprod weighting -> bf16 A matrix for GEMM2
    build_wnout<<<BB, 256, 0, stream>>>(out_nout, gate, wn);
    // 7) merged-K codeword GEMM -> output
    gemm2<<<dim3(BB / 128, CC / 128), 256, 0, stream>>>(wn, cwt, out_output);
    // 8) mask + exp reduction -> sfmx_base
    epilogue<<<BB, 256, 0, stream>>>(out_output, gate, nm, out_sfmx);
}
